// NeuralODE_63599875719302
// MI455X (gfx1250) — matrix operand order repaired and checked by reading, not compiled, not run on hardware
//
#include <hip/hip_runtime.h>
#include <hip/hip_bf16.h>
#include <math.h>

typedef __attribute__((ext_vector_type(16))) _Float16 v16h;
typedef __attribute__((ext_vector_type(8)))  _Float16 v8h;
typedef __attribute__((ext_vector_type(8)))  float    v8f;

// Dormand-Prince coefficients
#define A21f 0.161f
#define A31f (-0.008480655492356989f)
#define A32f 0.335480655492357f
#define A41f 2.8971530571054935f
#define A42f (-6.359448489975075f)
#define A43f 4.3622954328695815f
#define A51f 5.325864828439257f
#define A52f (-11.748883564062828f)
#define A53f 7.4955393428898365f
#define A54f (-0.09249506636175525f)
#define A61f 5.86145544294642f
#define A62f (-12.92096931784711f)
#define A63f 8.159367898576159f
#define A64f (-0.071584973281401f)
#define A65f (-0.028269050394068383f)
#define B1f 0.09646076681806523f
#define B2f 0.01f
#define B3f 0.4798896504144996f
#define B4f 1.379008574103742f
#define B5f (-3.290069515436081f)
#define B6f 2.324710524099774f
#define E1f (-0.001780011052225777f)
#define E2f (-0.0008164344596567469f)
#define E3f 0.007880878010261995f
#define E4f (-0.1447110071732629f)
#define E5f 0.5823571654525552f
#define E6f (-0.45808210592918697f)
#define E7f 0.015151515151515152f

#define MAX_STEPS 48
#define DT0f 0.1f
#define RTOLf 0.001f
#define ATOLf 1e-06f

#define LOG2Ef 1.44269504088896f
#define LN2f 0.69314718055995f

// LDS: W1h 16K + W2h 32K + W3h 16K + ybuf 2K + h1 4K + h2 4K + f32 tables
#define SMEM_BYTES (65536 + 2048 + 4096 + 4096 + (128 + 128 + 64) * 4 + 6 * 16 * 4)

// Raw HW transcendentals: v_exp_f32 is 2^x, v_log_f32 is log2(x).
#if __has_builtin(__builtin_amdgcn_exp2f) && __has_builtin(__builtin_amdgcn_logf)
__device__ __forceinline__ float fast_exp2(float x) { return __builtin_amdgcn_exp2f(x); }
__device__ __forceinline__ float fast_log2(float x) { return __builtin_amdgcn_logf(x); }
#else
__device__ __forceinline__ float fast_exp2(float x) { return exp2f(x); }
__device__ __forceinline__ float fast_log2(float x) { return __logf(x) * LOG2Ef; }
#endif

__device__ __forceinline__ v8f wmma_f16(v16h a, v16h b, v8f c) {
  // (neg_a, A, neg_b, B, c_mod, C, reuse_a, reuse_b)
  return __builtin_amdgcn_wmma_f32_16x16x32_f16(false, a, false, b, (short)0, c,
                                                false, false);
}

// A-fragment: 16x32 f16 tile from row-major LDS buffer (stride halves).
// lanes 0-15 -> row m, K[0..7]@V0-3, K[16..23]@V4-7;
// lanes16-31 -> row m, K[8..15]@V0-3, K[24..31]@V4-7.
__device__ __forceinline__ v16h load_a_frag(const _Float16* buf, int stride,
                                            int kt, int lane) {
  const int hf = lane >> 4, m = lane & 15;
  const _Float16* base = buf + m * stride + kt * 32;
  v8h lo = *(const v8h*)(base + hf * 8);
  v8h hi = *(const v8h*)(base + 16 + hf * 8);
  v16h a;
#pragma unroll
  for (int i = 0; i < 8; ++i) { a[i] = lo[i]; a[8 + i] = hi[i]; }
  return a;
}

// B-fragment: 32x16 f16 tile of W^T; column n of B == row n of W (row-major).
// lanes 0-15: K[0..7],K[16..23] of column n0+n; lanes 16-31: K[8..15],K[24..31].
__device__ __forceinline__ v16h load_b_frag(const _Float16* W, int stride,
                                            int nt, int kt, int lane) {
  const int hf = lane >> 4, n = lane & 15;
  const _Float16* base = W + (nt * 16 + n) * stride + kt * 32;
  v8h lo = *(const v8h*)(base + hf * 8);
  v8h hi = *(const v8h*)(base + 16 + hf * 8);
  v16h b;
#pragma unroll
  for (int i = 0; i < 8; ++i) { b[i] = lo[i]; b[8 + i] = hi[i]; }
  return b;
}

// Guard-free softplus: max(x,0) + ln2 * log2(1 + 2^(-|x|*log2e)).
// Argument of log2 is in (1,2] so no denormal handling is needed.
__device__ __forceinline__ v8f softplus8(v8f x) {
  v8f r;
#pragma unroll
  for (int j = 0; j < 8; ++j) {
    float v = x[j];
    float e = fast_exp2(fabsf(v) * -LOG2Ef);
    r[j] = fmaf(LN2f, fast_log2(1.0f + e), fmaxf(v, 0.0f));
  }
  return r;
}

// One MLP evaluation for the workgroup's 16-row batch tile.
// yin/kout: this wave's 16x16 column tile in WMMA C/D layout.
// Weight B-fragments are register-resident (bw1[q][kt], bw2[q][kt], bw3[kt]).
__device__ __forceinline__ void mlp_eval(
    const v8f& yin, v8f& kout,
    _Float16* ybuf, _Float16* h1buf, _Float16* h2buf,
    const v16h (&bw1)[2][2], const v16h (&bw2)[2][4], const v16h (&bw3)[4],
    const float* b1s, const float* b2s, const float* b3s,
    int wave, int lane) {
  const int hf = lane >> 4, n = lane & 15;
  // publish this wave's y tile (f16, row-major 16x64)
#pragma unroll
  for (int j = 0; j < 8; ++j)
    ybuf[(j + hf * 8) * 64 + wave * 16 + n] = (_Float16)yin[j];
  __syncthreads();

  // ---- Layer 1: (16x64) @ W1^T -> 16x128, softplus ----
  v8f h1t[2];
#pragma unroll
  for (int q = 0; q < 2; ++q) {
    const float bv = b1s[(wave * 2 + q) * 16 + n];
    v8f c;
#pragma unroll
    for (int j = 0; j < 8; ++j) c[j] = bv;
#pragma unroll
    for (int kt = 0; kt < 2; ++kt)
      c = wmma_f16(load_a_frag(ybuf, 64, kt, lane), bw1[q][kt], c);
    h1t[q] = softplus8(c);
  }
#pragma unroll
  for (int q = 0; q < 2; ++q) {
    const int nt = wave * 2 + q;
#pragma unroll
    for (int j = 0; j < 8; ++j)
      h1buf[(j + hf * 8) * 128 + nt * 16 + n] = (_Float16)h1t[q][j];
  }
  __syncthreads();

  // ---- Layer 2: (16x128) @ W2^T -> 16x128, softplus ----
  v8f h2t[2];
#pragma unroll
  for (int q = 0; q < 2; ++q) {
    const float bv = b2s[(wave * 2 + q) * 16 + n];
    v8f c;
#pragma unroll
    for (int j = 0; j < 8; ++j) c[j] = bv;
#pragma unroll
    for (int kt = 0; kt < 4; ++kt)
      c = wmma_f16(load_a_frag(h1buf, 128, kt, lane), bw2[q][kt], c);
    h2t[q] = softplus8(c);
  }
#pragma unroll
  for (int q = 0; q < 2; ++q) {
    const int nt = wave * 2 + q;
#pragma unroll
    for (int j = 0; j < 8; ++j)
      h2buf[(j + hf * 8) * 128 + nt * 16 + n] = (_Float16)h2t[q][j];
  }
  __syncthreads();

  // ---- Layer 3: (16x128) @ W3^T -> 16x64 (wave owns col-tile `wave`) ----
  {
    const float bv = b3s[wave * 16 + n];
    v8f c;
#pragma unroll
    for (int j = 0; j < 8; ++j) c[j] = bv;
#pragma unroll
    for (int kt = 0; kt < 4; ++kt)
      c = wmma_f16(load_a_frag(h2buf, 128, kt, lane), bw3[kt], c);
    kout = c;
  }
}

__global__ void __launch_bounds__(128)
ode_dopri5_kernel(const float* __restrict__ history,
                  const float* __restrict__ W1, const float* __restrict__ b1,
                  const float* __restrict__ W2, const float* __restrict__ b2,
                  const float* __restrict__ W3, const float* __restrict__ b3,
                  const int* __restrict__ steps_p, float* __restrict__ out) {
  extern __shared__ char smem[];
  _Float16* W1h   = (_Float16*)smem;       // 128*64
  _Float16* W2h   = W1h + 8192;            // 128*128
  _Float16* W3h   = W2h + 16384;           // 64*128
  _Float16* ybuf  = W3h + 8192;            // 16*64
  _Float16* h1buf = ybuf + 1024;           // 16*128
  _Float16* h2buf = h1buf + 2048;          // 16*128
  float* b1s    = (float*)(h2buf + 2048);
  float* b2s    = b1s + 128;
  float* b3s    = b2s + 128;
  float* t_s    = b3s + 64;
  float* dt_s   = t_s + 16;
  float* done_s = dt_s + 16;
  float* h_s    = done_s + 16;
  float* acc_s  = h_s + 16;
  float* es_s   = acc_s + 16;

  const int tid = threadIdx.x;
  const int wave = tid >> 5, lane = tid & 31;
  const int hf = lane >> 4, n = lane & 15;

  // Stage f16 weights + f32 biases into LDS
  for (int i = tid; i < 8192; i += 128) W1h[i] = (_Float16)W1[i];
  for (int i = tid; i < 16384; i += 128) W2h[i] = (_Float16)W2[i];
  for (int i = tid; i < 8192; i += 128) W3h[i] = (_Float16)W3[i];
  if (tid < 128) { b1s[tid] = b1[tid]; b2s[tid] = b2[tid]; }
  if (tid < 64) b3s[tid] = b3[tid];
  if (tid < 16) dt_s[tid] = DT0f;
  __syncthreads();

  // Hoist this wave's weight B-fragments into registers (live whole kernel).
  v16h bw1[2][2], bw2[2][4], bw3[4];
#pragma unroll
  for (int q = 0; q < 2; ++q) {
#pragma unroll
    for (int kt = 0; kt < 2; ++kt)
      bw1[q][kt] = load_b_frag(W1h, 64, wave * 2 + q, kt, lane);
#pragma unroll
    for (int kt = 0; kt < 4; ++kt)
      bw2[q][kt] = load_b_frag(W2h, 128, wave * 2 + q, kt, lane);
  }
#pragma unroll
  for (int kt = 0; kt < 4; ++kt)
    bw3[kt] = load_b_frag(W3h, 128, wave, kt, lane);

  const int b0 = blockIdx.x * 16;
  const int col = wave * 16 + n;
  const int nseg = steps_p[0];

  // y0 = history[:, -1]
  v8f y;
#pragma unroll
  for (int j = 0; j < 8; ++j) {
    const int m = j + hf * 8;
    y[j] = history[((size_t)(b0 + m) * 15 + 14) * 64 + col];
  }

  for (int seg = 0; seg < nseg; ++seg) {
    const float t1 = (float)(seg + 1);
    if (tid < 16) { t_s[tid] = t1 - 1.0f; done_s[tid] = 0.0f; }
    __syncthreads();

    for (int st = 0; st < MAX_STEPS; ++st) {
      if (tid < 16) {
        h_s[tid] = fminf(dt_s[tid], t1 - t_s[tid]);  // dt_c
        es_s[tid] = 0.0f;
      }
      __syncthreads();
      float hr[8];
#pragma unroll
      for (int j = 0; j < 8; ++j) hr[j] = h_s[j + hf * 8];

      v8f k1, k2, k3, k4, k5, k6, k7, tmp, y5;
      mlp_eval(y, k1, ybuf, h1buf, h2buf, bw1, bw2, bw3, b1s, b2s, b3s, wave, lane);
#pragma unroll
      for (int j = 0; j < 8; ++j)
        tmp[j] = fmaf(hr[j], A21f * k1[j], y[j]);
      mlp_eval(tmp, k2, ybuf, h1buf, h2buf, bw1, bw2, bw3, b1s, b2s, b3s, wave, lane);
#pragma unroll
      for (int j = 0; j < 8; ++j)
        tmp[j] = fmaf(hr[j], A31f * k1[j] + A32f * k2[j], y[j]);
      mlp_eval(tmp, k3, ybuf, h1buf, h2buf, bw1, bw2, bw3, b1s, b2s, b3s, wave, lane);
#pragma unroll
      for (int j = 0; j < 8; ++j)
        tmp[j] = fmaf(hr[j], A41f * k1[j] + A42f * k2[j] + A43f * k3[j], y[j]);
      mlp_eval(tmp, k4, ybuf, h1buf, h2buf, bw1, bw2, bw3, b1s, b2s, b3s, wave, lane);
#pragma unroll
      for (int j = 0; j < 8; ++j)
        tmp[j] = fmaf(hr[j],
                      A51f * k1[j] + A52f * k2[j] + A53f * k3[j] + A54f * k4[j],
                      y[j]);
      mlp_eval(tmp, k5, ybuf, h1buf, h2buf, bw1, bw2, bw3, b1s, b2s, b3s, wave, lane);
#pragma unroll
      for (int j = 0; j < 8; ++j)
        tmp[j] = fmaf(hr[j],
                      A61f * k1[j] + A62f * k2[j] + A63f * k3[j] +
                          A64f * k4[j] + A65f * k5[j],
                      y[j]);
      mlp_eval(tmp, k6, ybuf, h1buf, h2buf, bw1, bw2, bw3, b1s, b2s, b3s, wave, lane);
#pragma unroll
      for (int j = 0; j < 8; ++j)
        y5[j] = fmaf(hr[j],
                     B1f * k1[j] + B2f * k2[j] + B3f * k3[j] + B4f * k4[j] +
                         B5f * k5[j] + B6f * k6[j],
                     y[j]);
      mlp_eval(y5, k7, ybuf, h1buf, h2buf, bw1, bw2, bw3, b1s, b2s, b3s, wave, lane);

      // error norm: per-row RMS over 64 dims
#pragma unroll
      for (int j = 0; j < 8; ++j) {
        const float e = hr[j] * (E1f * k1[j] + E2f * k2[j] + E3f * k3[j] +
                                 E4f * k4[j] + E5f * k5[j] + E6f * k6[j] +
                                 E7f * k7[j]);
        const float sc = ATOLf + RTOLf * fmaxf(fabsf(y[j]), fabsf(y5[j]));
        const float r = e / sc;
        float s = r * r;
        s += __shfl_xor(s, 1, 32);
        s += __shfl_xor(s, 2, 32);
        s += __shfl_xor(s, 4, 32);
        s += __shfl_xor(s, 8, 32);
        if (n == 0) atomicAdd(&es_s[j + hf * 8], s);
      }
      __syncthreads();

      if (tid < 16) {  // per-row accept / step-size control
        const float en = sqrtf(es_s[tid] * (1.0f / 64.0f));
        const float dn = done_s[tid];
        const float dtc = h_s[tid];
        const bool acc = (en <= 1.0f) && (dn == 0.0f);
        // factor = 0.9 * en^-0.2, clamped to [0.2, 10]
        float factor =
            0.9f * fast_exp2(-0.2f * fast_log2(fmaxf(en, 1e-10f)));
        factor = fminf(fmaxf(factor, 0.2f), 10.0f);
        const float tn = acc ? (t_s[tid] + dtc) : t_s[tid];
        t_s[tid] = tn;
        acc_s[tid] = acc ? 1.0f : 0.0f;
        dt_s[tid] = (dn != 0.0f) ? dt_s[tid] : dtc * factor;
        done_s[tid] = ((dn != 0.0f) || (tn >= t1 - 1e-8f)) ? 1.0f : 0.0f;
      }
      __syncthreads();
#pragma unroll
      for (int j = 0; j < 8; ++j)
        if (acc_s[j + hf * 8] != 0.0f) y[j] = y5[j];
    }

    // emit ys[:, seg, :]
#pragma unroll
    for (int j = 0; j < 8; ++j) {
      const int m = j + hf * 8;
      out[((size_t)(b0 + m) * nseg + seg) * 64 + col] = y[j];
    }
  }
}

extern "C" void kernel_launch(void* const* d_in, const int* in_sizes, int n_in,
                              void* d_out, int out_size, void* d_ws,
                              size_t ws_size, hipStream_t stream) {
  (void)in_sizes; (void)n_in; (void)out_size; (void)d_ws; (void)ws_size;
  const float* history = (const float*)d_in[0];
  const float* W1 = (const float*)d_in[1];
  const float* b1 = (const float*)d_in[2];
  const float* W2 = (const float*)d_in[3];
  const float* b2 = (const float*)d_in[4];
  const float* W3 = (const float*)d_in[5];
  const float* b3 = (const float*)d_in[6];
  const int* steps = (const int*)d_in[7];
  float* out = (float*)d_out;

  hipFuncSetAttribute((const void*)ode_dopri5_kernel,
                      hipFuncAttributeMaxDynamicSharedMemorySize, SMEM_BYTES);

  // 2048 batch rows / 16 rows per workgroup; 4 waves (128 threads) cooperate
  // per 16-row tile by splitting the hidden dimension.
  ode_dopri5_kernel<<<dim3(128), dim3(128), SMEM_BYTES, stream>>>(
      history, W1, b1, W2, b2, W3, b3, steps, out);
}
